// MultiheadSelfAttention_10874857193993
// MI455X (gfx1250) — matrix operand order repaired and checked by reading, not compiled, not run on hardware
//
#include <hip/hip_runtime.h>

typedef __attribute__((ext_vector_type(16))) _Float16 v16h;
typedef __attribute__((ext_vector_type(8)))  _Float16 v8h;
typedef __attribute__((ext_vector_type(8)))  float    v8f;

#define NUM_HEADS 8
#define DIM_IN    512
#define DIM_QK    64
#define DV_H      64
#define SEQ       2048
#define BATCH     4
#define ROWS      (BATCH * SEQ)   /* 8192 */

// ---------------------------------------------------------------------------
// WMMA fragment loaders (layouts from CDNA5 ISA §7.12.2, wave32)
// ---------------------------------------------------------------------------

// A-fragment 16x32 f16 from row-major [*, lda].
static __device__ __forceinline__ v16h wmma_ld_a(const _Float16* A, int lda, int m0, int k0) {
  const int lane = threadIdx.x & 31;
  const int m  = m0 + (lane & 15);
  const int kh = k0 + ((lane >> 4) << 3);
  const _Float16* p = A + (size_t)m * lda + kh;
  v8h lo = *(const v8h*)(p);        // K = kh .. kh+7
  v8h hi = *(const v8h*)(p + 16);   // K = kh+16 .. kh+23
  v16h r;
#pragma unroll
  for (int i = 0; i < 8; ++i) { r[i] = lo[i]; r[8 + i] = hi[i]; }
  return r;
}

// B-fragment 32x16 f16. Bt is stored [N x K] row-major (column n of the
// logical KxN B-matrix is contiguous): lane half h holds K 8h..8h+7, 16+8h..16+8h+7.
static __device__ __forceinline__ v16h wmma_ld_b(const _Float16* Bt, int ldb, int n0, int k0) {
  const int lane = threadIdx.x & 31;
  const int n = n0 + (lane & 15);
  const int k = k0 + ((lane >> 4) << 3);
  const _Float16* p = Bt + (size_t)n * ldb + k;
  v8h lo = *(const v8h*)(p);
  v8h hi = *(const v8h*)(p + 16);
  v16h r;
#pragma unroll
  for (int i = 0; i < 8; ++i) { r[i] = lo[i]; r[8 + i] = hi[i]; }
  return r;
}

static __device__ __forceinline__ v8f wmma_f16(v16h a, v16h b, v8f c) {
  return __builtin_amdgcn_wmma_f32_16x16x32_f16(false, a, false, b, (short)0, c, false, false);
}

// Reductions across one 16-lane half (rows of a C tile live in one half).
static __device__ __forceinline__ float half_max(float v) {
#pragma unroll
  for (int m = 1; m <= 8; m <<= 1) v = fmaxf(v, __shfl_xor(v, m, 32));
  return v;
}
static __device__ __forceinline__ float half_sum(float v) {
#pragma unroll
  for (int m = 1; m <= 8; m <<= 1) v += __shfl_xor(v, m, 32);
  return v;
}

// ---------------------------------------------------------------------------
// CDNA5 async global->LDS DMA (ASYNCcnt-tracked). One instruction moves
// 16 B per lane = 512 B per wave. lds_off is the group-segment byte offset
// (low 32 bits of a flat shared pointer per the gfx1250 aperture rules).
// ---------------------------------------------------------------------------
static __device__ __forceinline__ unsigned lds_off_of(const void* p) {
  return (unsigned)(uintptr_t)p;
}
static __device__ __forceinline__ void async_copy_b128(unsigned lds_off, const void* gptr) {
  asm volatile("global_load_async_to_lds_b128 %0, %1, off"
               :: "v"(lds_off), "v"((unsigned long long)(uintptr_t)gptr)
               : "memory");
}

// ---------------------------------------------------------------------------
// Precision conversion / weight transpose
// ---------------------------------------------------------------------------
__global__ void cvt_kernel(const float* __restrict__ in, _Float16* __restrict__ out, int n) {
  int i = blockIdx.x * blockDim.x + threadIdx.x;
  int stride = gridDim.x * blockDim.x;
  for (; i < n; i += stride) out[i] = (_Float16)in[i];
}

// in[b][r][c] -> out[b][c][r]  (f32 -> f16)
__global__ void transpose_cvt_kernel(const float* __restrict__ in, _Float16* __restrict__ out,
                                     int batch, int R, int C) {
  int i = blockIdx.x * blockDim.x + threadIdx.x;
  int total = batch * R * C;
  int stride = gridDim.x * blockDim.x;
  for (; i < total; i += stride) {
    int b = i / (R * C), rem = i % (R * C);
    int r = rem / C, c = rem % C;
    out[(size_t)b * R * C + (size_t)c * R + r] = (_Float16)in[i];
  }
}

// ---------------------------------------------------------------------------
// QKV projections: [8192 x 512] x [512 x 64] per head, z selects Q/K/V.
// Q pre-scaled by 1/sqrt(64); V stored transposed [bh][dv][token].
// ---------------------------------------------------------------------------
__global__ __launch_bounds__(128) void qkv_kernel(
    const _Float16* __restrict__ featH,
    const _Float16* __restrict__ WqT, const _Float16* __restrict__ WkT,
    const _Float16* __restrict__ WvT,
    const float* __restrict__ bq, const float* __restrict__ bk, const float* __restrict__ bv,
    _Float16* __restrict__ Q, _Float16* __restrict__ K, _Float16* __restrict__ Vt) {
  const int wave = threadIdx.x >> 5;
  const int lane = threadIdx.x & 31;
  const int h = blockIdx.y;
  const int z = blockIdx.z;
  const int m0 = blockIdx.x * 64 + wave * 16;

  const _Float16* WT = (z == 0) ? WqT : (z == 1) ? WkT : WvT;
  const float* bias  = (z == 0) ? bq  : (z == 1) ? bk  : bv;
  WT += (size_t)h * DIM_QK * DIM_IN;

  v8f acc[4] = {};
#pragma unroll 4
  for (int k0 = 0; k0 < DIM_IN; k0 += 32) {
    v16h a = wmma_ld_a(featH, DIM_IN, m0, k0);
#pragma unroll
    for (int t = 0; t < 4; ++t) {
      v16h b = wmma_ld_b(WT, DIM_IN, t * 16, k0);
      acc[t] = wmma_f16(a, b, acc[t]);
    }
  }

  const float scale = (z == 0) ? 0.125f : 1.0f;
  const int bIdx = m0 / SEQ;
  const int nLoc = m0 % SEQ;
  const int c16  = lane & 15;
  const int roff = (lane >> 4) << 3;
#pragma unroll
  for (int t = 0; t < 4; ++t) {
    const int col = t * 16 + c16;
    const float bcol = bias[h * DIM_QK + col];
#pragma unroll
    for (int r = 0; r < 8; ++r) {
      float val = (acc[t][r] + bcol) * scale;
      int row = nLoc + roff + r;
      if (z == 2) {
        Vt[((size_t)(bIdx * NUM_HEADS + h) * DV_H + col) * SEQ + row] = (_Float16)val;
      } else {
        _Float16* dst = (z == 0) ? Q : K;
        dst[((size_t)(bIdx * NUM_HEADS + h) * SEQ + row) * DIM_QK + col] = (_Float16)val;
      }
    }
  }
}

// ---------------------------------------------------------------------------
// Flash attention with double-buffered async-DMA K/V staging in LDS.
// Block = 4 waves x 16 query rows; key blocks of 32 streamed via ASYNCcnt.
// ---------------------------------------------------------------------------
__global__ __launch_bounds__(128) void attn_kernel(
    const _Float16* __restrict__ Q, const _Float16* __restrict__ K,
    const _Float16* __restrict__ Vt, _Float16* __restrict__ attH) {
  // K block: 32 keys x 64 d (4 KB); V block: 64 dv x 32 tokens (4 KB); x2 buffers.
  __shared__ __attribute__((aligned(32))) _Float16 Ksh[2][32 * 64];
  __shared__ __attribute__((aligned(32))) _Float16 Vsh[2][64 * 32];
  __shared__ __attribute__((aligned(32))) _Float16 Psh[4][16 * 32];

  const int tid  = threadIdx.x;
  const int wave = tid >> 5;
  const int lane = tid & 31;
  const int bh = blockIdx.y;
  const int b = bh / NUM_HEADS, h = bh % NUM_HEADS;
  const int qLoc = blockIdx.x * 64 + wave * 16;

  const _Float16* Qp = Q  + (size_t)bh * SEQ * DIM_QK;
  const _Float16* Kp = K  + (size_t)bh * SEQ * DIM_QK;
  const _Float16* Vp = Vt + (size_t)bh * DV_H * SEQ;
  _Float16* pw = &Psh[wave][0];

  const v16h qa0 = wmma_ld_a(Qp, DIM_QK, qLoc, 0);
  const v16h qa1 = wmma_ld_a(Qp, DIM_QK, qLoc, 32);

  float mi[8], li[8];
#pragma unroll
  for (int r = 0; r < 8; ++r) { mi[r] = -1e30f; li[r] = 0.0f; }
  v8f o[4] = {};

  const int c16  = lane & 15;
  const int roff = (lane >> 4) << 3;

  // Stage a 32-key block: K rows [jb..jb+31][0..63] are 4 KB contiguous;
  // V rows Vt[dv][jb..jb+31] are 64 B each. 128 threads x 2 chunks x 16 B each.
  auto stage_kv = [&](int buf, int jb) {
#pragma unroll
    for (int i = 0; i < 2; ++i) {
      int c = tid + i * 128;                       // chunk 0..255
      async_copy_b128(lds_off_of(&Ksh[buf][0]) + (unsigned)c * 16,
                      Kp + (size_t)jb * DIM_QK + c * 8);
      int dv = c >> 2, part = c & 3;
      async_copy_b128(lds_off_of(&Vsh[buf][0]) + (unsigned)(dv * 64 + part * 16),
                      Vp + (size_t)dv * SEQ + jb + part * 8);
    }
  };

  stage_kv(0, 0);  // 4 async ops per wave in flight

  for (int jb = 0; jb < SEQ; jb += 32) {
    const int buf = (jb >> 5) & 1;
    if (jb + 32 < SEQ) {
      stage_kv(buf ^ 1, jb + 32);                          // prefetch next block
      asm volatile("s_wait_asynccnt 0x4" ::: "memory");    // current block's 4 ops done
    } else {
      asm volatile("s_wait_asynccnt 0x0" ::: "memory");
    }
    __syncthreads();  // all waves' DMA for this buffer landed

    // S = Q * K^T for 16 rows x 32 keys, K fragments from LDS
    v8f z = {};
    v16h kb = wmma_ld_b(&Ksh[buf][0], DIM_QK, 0, 0);
    v8f s0 = wmma_f16(qa0, kb, z);
    kb = wmma_ld_b(&Ksh[buf][0], DIM_QK, 0, 32);
    s0 = wmma_f16(qa1, kb, s0);
    kb = wmma_ld_b(&Ksh[buf][0], DIM_QK, 16, 0);
    v8f s1 = wmma_f16(qa0, kb, z);
    kb = wmma_ld_b(&Ksh[buf][0], DIM_QK, 16, 32);
    s1 = wmma_f16(qa1, kb, s1);

    // online softmax per row (row r+roff lives in one 16-lane half at elem r)
#pragma unroll
    for (int r = 0; r < 8; ++r) {
      float bm   = half_max(fmaxf(s0[r], s1[r]));
      float mnew = fmaxf(mi[r], bm);
      float p0 = __expf(s0[r] - mnew);
      float p1 = __expf(s1[r] - mnew);
      float rs = half_sum(p0 + p1);
      float corr = __expf(mi[r] - mnew);
      li[r] = li[r] * corr + rs;
      mi[r] = mnew;
#pragma unroll
      for (int t = 0; t < 4; ++t) o[t][r] *= corr;
      int row = roff + r;
      pw[row * 32 + c16]      = (_Float16)p0;
      pw[row * 32 + 16 + c16] = (_Float16)p1;
    }
    asm volatile("s_wait_dscnt 0x0" ::: "memory");  // same-wave LDS RAW on P tile

    // reload P as an A fragment (16x32) and accumulate O += P * V (V from LDS)
    {
      const int m  = lane & 15;
      const int kh = (lane >> 4) << 3;
      const _Float16* pp = pw + m * 32 + kh;
      v8h lo = *(const v8h*)(pp);
      v8h hi = *(const v8h*)(pp + 16);
      v16h pa;
#pragma unroll
      for (int i = 0; i < 8; ++i) { pa[i] = lo[i]; pa[8 + i] = hi[i]; }
#pragma unroll
      for (int t = 0; t < 4; ++t) {
        v16h vb = wmma_ld_b(&Vsh[buf][0], 32, t * 16, 0);
        o[t] = wmma_f16(pa, vb, o[t]);
      }
    }
    __syncthreads();  // everyone done reading this buffer before it is re-filled
  }

  float inv[8];
#pragma unroll
  for (int r = 0; r < 8; ++r) inv[r] = 1.0f / li[r];
#pragma unroll
  for (int t = 0; t < 4; ++t) {
#pragma unroll
    for (int r = 0; r < 8; ++r) {
      int grow = b * SEQ + qLoc + roff + r;
      attH[(size_t)grow * 512 + h * DV_H + t * 16 + c16] = (_Float16)(o[t][r] * inv[r]);
    }
  }
}

// ---------------------------------------------------------------------------
// Final projection: [8192 x 512] x [512 x 512] + bf, f32 output.
// ---------------------------------------------------------------------------
__global__ __launch_bounds__(128) void final_kernel(
    const _Float16* __restrict__ attH, const _Float16* __restrict__ WfT,
    const float* __restrict__ bf, float* __restrict__ out) {
  const int wave = threadIdx.x >> 5;
  const int lane = threadIdx.x & 31;
  const int m0 = blockIdx.x * 64 + wave * 16;
  const int nBase = blockIdx.y * 64;

  v8f acc[4] = {};
#pragma unroll 4
  for (int k0 = 0; k0 < 512; k0 += 32) {
    v16h a = wmma_ld_a(attH, 512, m0, k0);
#pragma unroll
    for (int t = 0; t < 4; ++t) {
      v16h bfr = wmma_ld_b(WfT, 512, nBase + t * 16, k0);
      acc[t] = wmma_f16(a, bfr, acc[t]);
    }
  }
  const int c16  = lane & 15;
  const int roff = (lane >> 4) << 3;
#pragma unroll
  for (int t = 0; t < 4; ++t) {
    const int col = nBase + t * 16 + c16;
    const float bias = bf[col];
#pragma unroll
    for (int r = 0; r < 8; ++r) {
      out[(size_t)(m0 + roff + r) * 512 + col] = acc[t][r] + bias;
    }
  }
}

// ---------------------------------------------------------------------------
extern "C" void kernel_launch(void* const* d_in, const int* in_sizes, int n_in,
                              void* d_out, int out_size, void* d_ws, size_t ws_size,
                              hipStream_t stream) {
  (void)in_sizes; (void)n_in; (void)out_size; (void)ws_size;
  const float* feat = (const float*)d_in[0];
  const float* Wq = (const float*)d_in[1];
  const float* bq = (const float*)d_in[2];
  const float* Wk = (const float*)d_in[3];
  const float* bk = (const float*)d_in[4];
  const float* Wv = (const float*)d_in[5];
  const float* bv = (const float*)d_in[6];
  const float* Wf = (const float*)d_in[7];
  const float* bf = (const float*)d_in[8];
  float* out = (float*)d_out;

  _Float16* ws = (_Float16*)d_ws;
  size_t off = 0;
  _Float16* featH = ws + off; off += (size_t)ROWS * DIM_IN;            // 8 MB
  _Float16* WqT   = ws + off; off += (size_t)NUM_HEADS * DIM_QK * DIM_IN;
  _Float16* WkT   = ws + off; off += (size_t)NUM_HEADS * DIM_QK * DIM_IN;
  _Float16* WvT   = ws + off; off += (size_t)NUM_HEADS * DV_H * DIM_IN;
  _Float16* WfT   = ws + off; off += (size_t)512 * 512;
  _Float16* Qh    = ws + off; off += (size_t)BATCH * NUM_HEADS * SEQ * DIM_QK;
  _Float16* Kh    = ws + off; off += (size_t)BATCH * NUM_HEADS * SEQ * DIM_QK;
  _Float16* Vth   = ws + off; off += (size_t)BATCH * NUM_HEADS * DV_H * SEQ;
  _Float16* attH  = featH;   // featH dead after QKV; reuse for attention output

  cvt_kernel<<<1024, 256, 0, stream>>>(feat, featH, ROWS * DIM_IN);
  transpose_cvt_kernel<<<512, 256, 0, stream>>>(Wq, WqT, NUM_HEADS, DIM_IN, DIM_QK);
  transpose_cvt_kernel<<<512, 256, 0, stream>>>(Wk, WkT, NUM_HEADS, DIM_IN, DIM_QK);
  transpose_cvt_kernel<<<512, 256, 0, stream>>>(Wv, WvT, NUM_HEADS, DIM_IN, DV_H);
  transpose_cvt_kernel<<<512, 256, 0, stream>>>(Wf, WfT, 1, 512, 512);

  qkv_kernel<<<dim3(ROWS / 64, NUM_HEADS, 3), 128, 0, stream>>>(
      featH, WqT, WkT, WvT, bq, bk, bv, Qh, Kh, Vth);
  attn_kernel<<<dim3(SEQ / 64, BATCH * NUM_HEADS), 128, 0, stream>>>(Qh, Kh, Vth, attH);
  final_kernel<<<dim3(ROWS / 64, 512 / 64), 128, 0, stream>>>(attH, WfT, bf, out);
}
